// Modelv8_28114855919777
// MI455X (gfx1250) — compile-verified, matrix operand order checked by reading, not run on hardware
//
#include <hip/hip_runtime.h>
#include <hip/hip_bf16.h>

// ---------------------------------------------------------------------------
// CDNA5 (gfx1250) 2-layer heterogeneous GAT.
// Dense projections: bf16 pre-packed operands + v_wmma_f32_16x16x32_bf16,
// 16x64 output per wave, fully unrolled K loop, b128 operand loads.
// Edge softmax/aggregation: atomic scatter path (bandwidth bound).
// ---------------------------------------------------------------------------

#define N_OER 30000
#define N_CON 15000
#define N_CLS 1500
#define FEAT  128
#define HID   64
#define HEADS 4
#define HC    256
#define E_SR  60000
#define E_EP  250000
#define E_COV 250000
#define E_BEL 40000

typedef __attribute__((ext_vector_type(16))) __bf16 v16bf;
typedef __attribute__((ext_vector_type(8)))  __bf16 v8bf;
typedef __attribute__((ext_vector_type(8)))  float  v8f;

static __device__ __forceinline__ unsigned short f2bfbits(float f) {
    unsigned u = __float_as_uint(f);
    unsigned r = u + 0x7fffu + ((u >> 16) & 1u);   // round-to-nearest-even
    return (unsigned short)(r >> 16);
}

static __device__ __forceinline__ v8bf ld_v8bf(const unsigned short* p) {
    union { uint4 u; v8bf b; } cv;
    cv.u = *(const uint4*)p;                        // global_load_b128
    return cv.b;
}
static __device__ __forceinline__ v16bf cat16(v8bf lo, v8bf hi) {
    return __builtin_shufflevector(lo, hi, 0, 1, 2, 3, 4, 5, 6, 7,
                                           8, 9, 10, 11, 12, 13, 14, 15);
}

// monotonic float<->uint encoding for atomicMax-based segment max
static __device__ __forceinline__ unsigned encf(float v) {
    unsigned u = __float_as_uint(v);
    return (u & 0x80000000u) ? ~u : (u | 0x80000000u);
}
static __device__ __forceinline__ float decf(unsigned e) {
    return __uint_as_float((e & 0x80000000u) ? (e ^ 0x80000000u) : ~e);
}
#define ENC_NEG_INF 0x007FFFFFu   // encf(-inf)

// ---------------------------------------------------------------------------
// Operand packing.
// X -> bf16, row-major (A fragment = two contiguous b128 runs per K chunk).
// W -> bf16 swizzled: Wp[((chunk*(out/16)+coltile)*32 + lane)*16 + 2v {,+1}]
//   holds W[k, col], W[k+1, col] with k = chunk*32 + (v<4 ? g*8+2v
//   : 16+g*8+2(v-4)), col = coltile*16 + lane%16  -> one lane = 32 contig B.
// ---------------------------------------------------------------------------
__global__ void pack_x_kernel(const float* __restrict__ X,
                              unsigned short* __restrict__ Xb, long total)
{
    const long idx = (long)blockIdx.x * blockDim.x + threadIdx.x;
    if (idx < total) Xb[idx] = f2bfbits(X[idx]);
}

__global__ void pack_w_kernel(const float* __restrict__ W,
                              unsigned short* __restrict__ Wp,
                              int in_dim, int out_dim)
{
    const int idx = blockIdx.x * blockDim.x + threadIdx.x;
    const int total = (in_dim / 32) * (out_dim / 16) * 32 * 8;
    if (idx >= total) return;
    const int v    = idx & 7;
    const int lane = (idx >> 3) & 31;
    const int rest = idx >> 8;
    const int nct  = out_dim >> 4;
    const int ct   = rest % nct;
    const int ch   = rest / nct;
    const int g    = lane >> 4;
    const int k    = ch * 32 + ((v < 4) ? (g * 8 + 2 * v)
                                        : (16 + g * 8 + 2 * (v - 4)));
    const int col  = ct * 16 + (lane & 15);
    Wp[(size_t)idx * 2]     = f2bfbits(W[(size_t)k       * out_dim + col]);
    Wp[(size_t)idx * 2 + 1] = f2bfbits(W[(size_t)(k + 1) * out_dim + col]);
}

// Y[n x out] = X @ W (+ bias). One wave -> 16 rows x 64 cols (4 subtiles).
template <int IN_DIM>
__global__ void wmma_linear_kernel(const unsigned short* __restrict__ Xb,
                                   const unsigned short* __restrict__ Wp,
                                   const float* __restrict__ bias,
                                   float* __restrict__ Y,
                                   int n, int out_dim)
{
    const int lane = threadIdx.x & 31;
    const int g    = lane >> 4;
    const int ml   = lane & 15;
    const int row  = blockIdx.x * 16 + ml;
    const int rowc = (row < n) ? row : 0;        // clamp: padded rows unstored
    const int ct0  = blockIdx.y * 4;
    const int nct  = out_dim >> 4;

    const unsigned short* xrow = Xb + (size_t)rowc * IN_DIM;
    v8f c0 = {}, c1 = {}, c2 = {}, c3 = {};

#pragma unroll
    for (int ch = 0; ch < IN_DIM / 32; ++ch) {
        const v16bf a = cat16(ld_v8bf(xrow + ch * 32 + g * 8),
                              ld_v8bf(xrow + ch * 32 + 16 + g * 8));
#pragma unroll
        for (int nt = 0; nt < 4; ++nt) {
            const unsigned short* bp =
                Wp + ((size_t)(ch * nct + ct0 + nt) * 32 + lane) * 16;
            const v16bf b = cat16(ld_v8bf(bp), ld_v8bf(bp + 8));
            v8f& c = (nt == 0) ? c0 : (nt == 1) ? c1 : (nt == 2) ? c2 : c3;
            c = __builtin_amdgcn_wmma_f32_16x16x32_bf16(
                    false, a, false, b, (short)0, c, false, false);
        }
    }

#pragma unroll
    for (int nt = 0; nt < 4; ++nt) {
        const int col = (ct0 + nt) * 16 + ml;
        const float bv = bias ? bias[col] : 0.f;
        const v8f& c = (nt == 0) ? c0 : (nt == 1) ? c1 : (nt == 2) ? c2 : c3;
#pragma unroll
        for (int r = 0; r < 8; ++r) {
            const int orow = blockIdx.x * 16 + r + g * 8;
            if (orow < n) Y[(size_t)orow * out_dim + col] = c[r] + bv;
        }
    }
}

// a[n,4] = sum_k h[n, hd*64+k] * att[hd,k]   (h is n x 256)
__global__ void att_reduce_kernel(const float* __restrict__ h,
                                  const float* __restrict__ att,
                                  float* __restrict__ a, int n)
{
    const int idx = blockIdx.x * blockDim.x + threadIdx.x;
    if (idx >= n * HEADS) return;
    const int node = idx >> 2, hd = idx & 3;
    const float* hp = h + (size_t)node * HC + hd * HID;
    const float* ap = att + hd * HID;
    float s = 0.f;
#pragma unroll 8
    for (int k = 0; k < HID; ++k) s += hp[k] * ap[k];
    a[idx] = s;
}

// wt[in,4] = sum_k W[in, hd*64+k] * att[hd,k]   (fold att into W_dst)
__global__ void fold_att_kernel(const float* __restrict__ W,
                                const float* __restrict__ att,
                                float* __restrict__ wt, int in_dim)
{
    const int idx = blockIdx.x * blockDim.x + threadIdx.x;
    if (idx >= in_dim * HEADS) return;
    const int in = idx >> 2, hd = idx & 3;
    float s = 0.f;
#pragma unroll 8
    for (int k = 0; k < HID; ++k)
        s += W[(size_t)in * HC + hd * HID + k] * att[hd * HID + k];
    wt[idx] = s;
}

// a[n,4] = X[n, :in_dim] @ wt[:,4]
__global__ void gemv4_kernel(const float* __restrict__ X,
                             const float* __restrict__ wt,
                             float* __restrict__ a, int n, int in_dim)
{
    const int idx = blockIdx.x * blockDim.x + threadIdx.x;
    if (idx >= n * HEADS) return;
    const int node = idx >> 2, hd = idx & 3;
    const float* xp = X + (size_t)node * in_dim;
    float s = 0.f;
    for (int k = 0; k < in_dim; ++k) s += xp[k] * wt[k * 4 + hd];
    a[idx] = s;
}

__global__ void gat_init_kernel(float* __restrict__ acc,
                                float* __restrict__ den,
                                unsigned* __restrict__ menc, int n_dst)
{
    const long idx = (long)blockIdx.x * blockDim.x + threadIdx.x;
    if (idx < (long)n_dst * HC) acc[idx] = 0.f;
    if (idx < (long)n_dst * HEADS) { den[idx] = 0.f; menc[idx] = ENC_NEG_INF; }
}

__global__ void edge_logit_kernel(const int* __restrict__ src,
                                  const int* __restrict__ dst,
                                  int E, int nself,
                                  const float* __restrict__ aS,
                                  const float* __restrict__ aD,
                                  float* __restrict__ elog,
                                  unsigned* __restrict__ menc)
{
    const long idx = (long)blockIdx.x * blockDim.x + threadIdx.x;
    if (idx >= (long)(E + nself) * HEADS) return;
    const int e = (int)(idx >> 2), hd = (int)(idx & 3);
    int s, d;
    if (e < E) { s = src[e]; d = dst[e]; } else { s = d = e - E; }
    float v = aS[s * 4 + hd] + aD[d * 4 + hd];
    v = (v < 0.f) ? 0.2f * v : v;          // leaky_relu(.., 0.2)
    elog[idx] = v;
    atomicMax(&menc[d * 4 + hd], encf(v));
}

__global__ void edge_exp_kernel(const int* __restrict__ src,
                                const int* __restrict__ dst,
                                int E, int nself,
                                const float* __restrict__ elog,
                                const unsigned* __restrict__ menc,
                                float* __restrict__ eexp,
                                float* __restrict__ den)
{
    const long idx = (long)blockIdx.x * blockDim.x + threadIdx.x;
    if (idx >= (long)(E + nself) * HEADS) return;
    const int e = (int)(idx >> 2), hd = (int)(idx & 3);
    int d;
    if (e < E) { d = dst[e]; } else { d = e - E; }
    const float ex = __expf(elog[idx] - decf(menc[d * 4 + hd]));
    eexp[idx] = ex;
    atomicAdd(&den[d * 4 + hd], ex);
}

// one block (256 threads) per edge: acc[dst, k] += h_src[src, k] * alpha
__global__ void edge_aggr_kernel(const int* __restrict__ src,
                                 const int* __restrict__ dst,
                                 int E,
                                 const float* __restrict__ pS,
                                 const float* __restrict__ eexp,
                                 const float* __restrict__ den,
                                 float* __restrict__ acc)
{
    const int e = blockIdx.x;
    const int k = threadIdx.x;          // 0..255
    int s, d;
    if (e < E) { s = src[e]; d = dst[e]; } else { s = d = e - E; }
    const int hd = k >> 6;
    const float alpha = eexp[(long)e * 4 + hd] / (den[d * 4 + hd] + 1e-16f);
    atomicAdd(&acc[(size_t)d * HC + k], pS[(size_t)s * HC + k] * alpha);
}

// dst = (accumulate ? dst : 0) + scale * (acc + bias)
__global__ void combine_kernel(float* __restrict__ dst,
                               const float* __restrict__ acc,
                               const float* __restrict__ bias,
                               int n, float scale, int accumulate)
{
    const long idx = (long)blockIdx.x * blockDim.x + threadIdx.x;
    if (idx >= (long)n * HC) return;
    const int col = (int)(idx & (HC - 1));
    const float v = scale * (acc[idx] + bias[col]);
    dst[idx] = accumulate ? dst[idx] + v : v;
}

// out[e] = concat(xO[s], hO[s], xO[d], hO[d]) @ W + b ; wave per edge
__global__ void classifier_kernel(const int* __restrict__ src,
                                  const int* __restrict__ dst, int E,
                                  const float* __restrict__ xO,
                                  const float* __restrict__ hO,
                                  const float* __restrict__ W,
                                  const float* __restrict__ b,
                                  float* __restrict__ out)
{
    const int wid  = (blockIdx.x * blockDim.x + threadIdx.x) >> 5;
    const int lane = threadIdx.x & 31;
    if (wid >= E) return;
    const int s = src[wid], d = dst[wid];
    float sum = 0.f;
    for (int k = lane; k < FEAT; k += 32)
        sum += xO[(size_t)s * FEAT + k] * W[k]
             + xO[(size_t)d * FEAT + k] * W[384 + k];
    for (int k = lane; k < HC; k += 32)
        sum += hO[(size_t)s * HC + k] * W[FEAT + k]
             + hO[(size_t)d * HC + k] * W[512 + k];
#pragma unroll
    for (int off = 16; off; off >>= 1) sum += __shfl_xor(sum, off, 32);
    if (lane == 0) out[wid] = sum + b[0];
}

// ---------------------------------------------------------------------------
extern "C" void kernel_launch(void* const* d_in, const int* in_sizes, int n_in,
                              void* d_out, int out_size, void* d_ws, size_t ws_size,
                              hipStream_t stream)
{
    (void)in_sizes; (void)n_in; (void)out_size; (void)ws_size;
    auto F = [&](int i) { return (const float*)d_in[i]; };
    auto I = [&](int i) { return (const int*)d_in[i]; };

    // ---- input layout (setup_inputs dict order; params in jax tree_flatten
    //      sorted-key order: cls{W,b} < layers[0..1]{before_ep,belongs,covers,
    //      rev_belongs,rev_covers}{W_dst,W_src,att_dst,att_src,bias} < lin) ---
    const float* x_oer = F(0);
    const float* x_con = F(1);
    const float* x_cls = F(2);
    const float* clsW  = F(3);
    const float* clsb  = F(4);
    // layer l, edge-type t: leaf base = 5 + l*25 + t*5
    // t: 0=before_ep 1=belongs 2=covers 3=rev_belongs 4=rev_covers
    const float* linW[3] = { F(55), F(57), F(59) };  // Class, Concept, OER
    const float* linB[3] = { F(56), F(58), F(60) };
    const int* eiSR  = I(61);
    const int* eiEP  = I(62);
    const int* eiCOV = I(63);
    const int* eiBEL = I(64);
    const int* eiRCV = I(65);
    const int* eiRBL = I(66);

    // ---- workspace layout ----
    float* ws = (float*)d_ws;
    size_t o = 0;
    float* hOa = ws + o; o += (size_t)N_OER * HC;
    float* hCa = ws + o; o += (size_t)N_CON * HC;
    float* hKa = ws + o; o += (size_t)N_CLS * HC;
    float* hOb = ws + o; o += (size_t)N_OER * HC;
    float* hCb = ws + o; o += (size_t)N_CON * HC;
    float* hKb = ws + o; o += (size_t)N_CLS * HC;
    float* pS  = ws + o; o += (size_t)N_OER * HC;       // projected src feats
    float* acc = ws + o; o += (size_t)N_OER * HC;       // per-GAT accumulator
    float* aS  = ws + o; o += (size_t)N_OER * HEADS;
    float* aD  = ws + o; o += (size_t)N_OER * HEADS;
    unsigned* menc = (unsigned*)(ws + o); o += (size_t)N_OER * HEADS;
    float* den = ws + o; o += (size_t)N_OER * HEADS;
    float* elog= ws + o; o += (size_t)(E_EP + N_OER) * HEADS;
    float* eexp= ws + o; o += (size_t)(E_EP + N_OER) * HEADS;
    float* wt  = ws + o; o += (size_t)HC * HEADS;
    unsigned short* Xbf = (unsigned short*)(ws + o); o += (size_t)N_OER * HC / 2;
    unsigned short* Wp  = (unsigned short*)(ws + o); o += (size_t)HC * HC / 2;

    // ---- dense linear via packed-bf16 WMMA ----
    auto linear = [&](const float* X, const float* W, const float* bias,
                      float* Y, int n, int in_dim, int out_dim) {
        const long xt = (long)n * in_dim;
        pack_x_kernel<<<(int)((xt + 255) / 256), 256, 0, stream>>>(X, Xbf, xt);
        const int wtot = (in_dim / 32) * (out_dim / 16) * 32 * 8;
        pack_w_kernel<<<(wtot + 255) / 256, 256, 0, stream>>>(W, Wp, in_dim, out_dim);
        dim3 grid((n + 15) / 16, out_dim / 64);
        if (in_dim == 64)
            wmma_linear_kernel<64><<<grid, 32, 0, stream>>>(Xbf, Wp, bias, Y, n, out_dim);
        else if (in_dim == 128)
            wmma_linear_kernel<128><<<grid, 32, 0, stream>>>(Xbf, Wp, bias, Y, n, out_dim);
        else
            wmma_linear_kernel<256><<<grid, 32, 0, stream>>>(Xbf, Wp, bias, Y, n, out_dim);
    };

    // ---- initial per-type linear: h = x @ W(128x64) + b ----
    linear(x_oer, linW[2], linB[2], hOa, N_OER, FEAT, HID);
    linear(x_con, linW[1], linB[1], hCa, N_CON, FEAT, HID);
    linear(x_cls, linW[0], linB[0], hKa, N_CLS, FEAT, HID);

    // one GAT pass: fills `acc` (n_dst x 256); caller combines with bias
    auto gat = [&](const float* xs, int ns, const float* xd, int nd, int in_dim,
                   const int* ei, int E, int nself, int base) {
        const float* Wd = F(base + 0);
        const float* Wsr= F(base + 1);
        const float* ad = F(base + 2);
        const float* as = F(base + 3);
        const int* srcI = ei;            // (2,E) row-major
        const int* dstI = ei + E;
        const int Etot  = E + nself;

        // h_src projection (WMMA), then a_src from it
        linear(xs, Wsr, nullptr, pS, ns, in_dim, HC);
        att_reduce_kernel<<<(ns * HEADS + 255) / 256, 256, 0, stream>>>(pS, as, aS, ns);

        // a_dst via folded W_dst*att_dst (never materialize h_dst)
        fold_att_kernel<<<(in_dim * HEADS + 255) / 256, 256, 0, stream>>>(Wd, ad, wt, in_dim);
        gemv4_kernel<<<(nd * HEADS + 255) / 256, 256, 0, stream>>>(xd, wt, aD, nd, in_dim);

        gat_init_kernel<<<(int)(((long)nd * HC + 255) / 256), 256, 0, stream>>>(acc, den, menc, nd);

        const int blks4 = (int)(((long)Etot * HEADS + 255) / 256);
        edge_logit_kernel<<<blks4, 256, 0, stream>>>(srcI, dstI, E, nself, aS, aD, elog, menc);
        edge_exp_kernel  <<<blks4, 256, 0, stream>>>(srcI, dstI, E, nself, elog, menc, eexp, den);
        edge_aggr_kernel <<<Etot, HC, 0, stream>>>(srcI, dstI, E, pS, eexp, den, acc);
    };
    auto combine = [&](float* dst, int n, int base, float scale, int accum) {
        const float* bias = F(base + 4);
        combine_kernel<<<(int)(((long)n * HC + 255) / 256), 256, 0, stream>>>(
            dst, acc, bias, n, scale, accum);
    };

    // ---- 2 GAT layers ----
    float *hO = hOa, *hC = hCa, *hK = hKa;
    float *nO = hOb, *nC = hCb, *nK = hKb;
    int ind = HID;
    for (int l = 0; l < 2; ++l) {
        const int base = 5 + l * 25;
        // OER <- 0.5*(before_ep(+self loops) + rev_covers)
        gat(hO, N_OER, hO, N_OER, ind, eiEP, E_EP, N_OER, base + 0 * 5);
        combine(nO, N_OER, base + 0 * 5, 0.5f, 0);
        gat(hC, N_CON, hO, N_OER, ind, eiRCV, E_COV, 0, base + 4 * 5);
        combine(nO, N_OER, base + 4 * 5, 0.5f, 1);
        // Concept <- 0.5*(covers + rev_belongs)
        gat(hO, N_OER, hC, N_CON, ind, eiCOV, E_COV, 0, base + 2 * 5);
        combine(nC, N_CON, base + 2 * 5, 0.5f, 0);
        gat(hK, N_CLS, hC, N_CON, ind, eiRBL, E_BEL, 0, base + 3 * 5);
        combine(nC, N_CON, base + 3 * 5, 0.5f, 1);
        // Class <- belongs
        gat(hC, N_CON, hK, N_CLS, ind, eiBEL, E_BEL, 0, base + 1 * 5);
        combine(nK, N_CLS, base + 1 * 5, 1.0f, 0);

        float* t;
        t = hO; hO = nO; nO = t;
        t = hC; hC = nC; nC = t;
        t = hK; hK = nK; nK = t;
        ind = HC;
    }

    // ---- edge classifier over ei_before_sr ----
    classifier_kernel<<<(E_SR * 32 + 255) / 256, 256, 0, stream>>>(
        eiSR, eiSR + E_SR, E_SR, x_oer, hO, clsW, clsb, (float*)d_out);
}
